// FTransformer_49649821942377
// MI455X (gfx1250) — matrix operand order repaired and checked by reading, not compiled, not run on hardware
//
#include <hip/hip_runtime.h>
#include <hip/hip_bf16.h>
#include <math.h>

typedef __attribute__((ext_vector_type(16))) _Float16 v16h;
typedef __attribute__((ext_vector_type(8)))  float    v8f;

// ---------------- WMMA fragment helpers (contiguous k order: k = 16*half + e) ----------------

// A-matrix 16x32 f16, row-major source, row stride lda (elements).
// lane m = lane&15 (row), half = lane>>4.  element e -> k = 16*half + e
__device__ inline v16h load_a_f16(const _Float16* base, int lda, int lane) {
  const int half = (lane >> 4) & 1;
  const int m    = lane & 15;
  const _Float16* p = base + (size_t)m * lda + half * 16;
  v16h a;
#pragma unroll
  for (int e = 0; e < 16; ++e) {
    int k = e;
    a[e] = p[k];
  }
  return a;
}

// B-matrix 32x16 f16 from row-major K x N source (ldb = N).
// lane n = lane&15 (col), khalf = lane>>4.  element e -> k = 16*khalf + e
__device__ inline v16h load_b_rm(const _Float16* base, int ldb, int lane) {
  const int kh = (lane >> 4) & 1;
  const int n  = lane & 15;
  const _Float16* p = base + (size_t)(kh * 16) * ldb + n;
  v16h b;
#pragma unroll
  for (int e = 0; e < 16; ++e) b[e] = p[(size_t)e * ldb];
  return b;
}

// B-matrix 32x16 f16 where B[k][n] = M[n][k] (M row-major N x K, ld = row stride).
// Each lane reads 16 contiguous f16.
__device__ inline v16h load_b_tr(const _Float16* base, int ld, int lane) {
  const int kh = (lane >> 4) & 1;
  const int n  = lane & 15;
  const _Float16* p = base + (size_t)n * ld + kh * 16;
  v16h b;
#pragma unroll
  for (int e = 0; e < 16; ++e) b[e] = p[e];
  return b;
}

__device__ inline v8f wmma_f16(v16h a, v16h b, v8f c) {
  return __builtin_amdgcn_wmma_f32_16x16x32_f16(false, a, false, b, (short)0, c, false, false);
}

__device__ inline float silu_f(float x) { return x / (1.0f + expf(-x)); }

// laplace: 0.5*(1+erf((x-sqrt(0.5))/sqrt(0.5*pi)))
__device__ inline float laplace_f(float x) {
  return 0.5f * (1.0f + erff((x - 0.70710678118654752f) * 0.79788456080286536f));
}

#define BATCH   128
#define NSEQ    1800
#define NPAD    2048
#define NGRP    8
#define GRP     256
#define DMODEL  6
#define EDIM    12
#define QKD     128
#define HSTRIDE 10816   // 10800 padded to multiple of 32

// ---------------- 1. per-token preprocessing ----------------
__global__ void k_preproc(const float* __restrict__ x,
                          const float* __restrict__ ln_g, const float* __restrict__ ln_b,
                          const float* __restrict__ Wh,   const float* __restrict__ bh,
                          const float* __restrict__ Wqk,  const float* __restrict__ bqk,
                          const float* __restrict__ osg,  const float* __restrict__ osb,
                          _Float16* __restrict__ qq, _Float16* __restrict__ lq,
                          _Float16* __restrict__ qk2, _Float16* __restrict__ lk,
                          _Float16* __restrict__ vbuf, _Float16* __restrict__ gate) {
  const int T = blockIdx.x;            // token id in padded space
  const int b = T >> 11;
  const int np = T & (NPAD - 1);
  const int t = threadIdx.x;           // 0..127
  const size_t tok = (size_t)T;

  if (np >= NSEQ) {                    // padded token: zero everything
    qq [tok * QKD + t] = (_Float16)0.0f;
    lq [tok * QKD + t] = (_Float16)0.0f;
    qk2[tok * QKD + t] = (_Float16)0.0f;
    lk [tok * QKD + t] = (_Float16)0.0f;
    if (t < 16) vbuf[tok * 16 + t] = (_Float16)0.0f;
    return;
  }

  const float* xp = x + ((size_t)b * NSEQ + np) * DMODEL;
  float xv[DMODEL];
  float mu = 0.f;
#pragma unroll
  for (int d = 0; d < DMODEL; ++d) { xv[d] = xp[d]; mu += xv[d]; }
  mu *= (1.0f / DMODEL);
  float var = 0.f;
#pragma unroll
  for (int d = 0; d < DMODEL; ++d) { float df = xv[d] - mu; var += df * df; }
  var *= (1.0f / DMODEL);
  const float rstd = rsqrtf(var + 1e-5f);
  float nr[DMODEL];
#pragma unroll
  for (int d = 0; d < DMODEL; ++d) nr[d] = (xv[d] - mu) * rstd * ln_g[d] + ln_b[d];

  // qk head (all 128 threads)
  float s = bqk[t];
#pragma unroll
  for (int d = 0; d < DMODEL; ++d) s += nr[d] * Wqk[d * QKD + t];
  const float q = silu_f(s);
  qq [tok * QKD + t] = (_Float16)(q * osg[0 * QKD + t] + osb[0 * QKD + t]);
  lq [tok * QKD + t] = (_Float16)(q * osg[1 * QKD + t] + osb[1 * QKD + t]);
  qk2[tok * QKD + t] = (_Float16)(q * osg[2 * QKD + t] + osb[2 * QKD + t]);
  lk [tok * QKD + t] = (_Float16)(q * osg[3 * QKD + t] + osb[3 * QKD + t]);

  if (t < 24) {
    float s2 = bh[t];
#pragma unroll
    for (int d = 0; d < DMODEL; ++d) s2 += nr[d] * Wh[d * 24 + t];
    const float hv = silu_f(s2);
    if (t < EDIM) vbuf[tok * 16 + t] = (_Float16)hv;
    else          gate[((size_t)b * NSEQ + np) * EDIM + (t - EDIM)] = (_Float16)hv;
  }
  if (t >= 24 && t < 28) vbuf[tok * 16 + (t - 12)] = (_Float16)0.0f; // cols 12..15
}

// ---------------- 2a. per-group kv = lk^T v / 256 (WMMA) ----------------
// 8 waves/block; wave w computes the 16x16 tile d = w*16..w*16+15, e = 0..15 (K = 256)
__global__ void __launch_bounds__(256)
k_kv(const _Float16* __restrict__ lk, const _Float16* __restrict__ vbuf,
     float* __restrict__ kvr) {
  const int bg = blockIdx.x;                     // b*8+g
  const size_t tokbase = (size_t)bg * GRP;
  const int wave = threadIdx.x >> 5;             // d-tile
  const int lane = threadIdx.x & 31;
  const int half = (lane >> 4) & 1;
  const int m = lane & 15;

  v8f c = {};
  for (int k0 = 0; k0 < GRP; k0 += 32) {
    // A[m][k] = lk[(tokbase + k0 + k)*QKD + wave*16 + m]  (column-major gather)
    const _Float16* P = lk + (tokbase + k0) * QKD + wave * 16 + m;
    v16h a;
#pragma unroll
    for (int e = 0; e < 16; ++e) {
      int kk = e + half * 16;
      a[e] = P[(size_t)kk * QKD];
    }
    v16h bb = load_b_rm(vbuf + (tokbase + k0) * 16, 16, lane);
    c = wmma_f16(a, bb, c);
  }
  const int e = lane & 15;
  if (e < EDIM) {
    const int dbase = wave * 16 + half * 8;
#pragma unroll
    for (int r = 0; r < 8; ++r)
      kvr[((size_t)bg * QKD + dbase + r) * EDIM + e] = c[r] * (1.0f / GRP);
  }
}

// ---------------- 2b. exclusive cumsum over groups ----------------
__global__ void k_cumsum(const float* __restrict__ kvr, float* __restrict__ kvc) {
  const int b = blockIdx.x;
  for (int p = threadIdx.x; p < QKD * EDIM; p += blockDim.x) {
    float run = 0.f;
    for (int g = 0; g < NGRP; ++g) {
      const size_t idx = ((size_t)(b * NGRP + g)) * (QKD * EDIM) + p;
      kvc[idx] = run;
      run += kvr[idx];
    }
  }
}

// ---------------- 3. fused attention (quad + lin) + output proj ----------------
__global__ void __launch_bounds__(512)
k_attn(const _Float16* __restrict__ qq, const _Float16* __restrict__ qk2,
       const _Float16* __restrict__ lq, const _Float16* __restrict__ vbuf,
       const _Float16* __restrict__ gate, const float* __restrict__ kvc,
       const float* __restrict__ rel, const float* __restrict__ x,
       const float* __restrict__ Wo, const float* __restrict__ bo,
       _Float16* __restrict__ h) {
  const int bg = blockIdx.x;
  const int b = bg >> 3, g = bg & 7;
  const size_t tokbase = (size_t)bg * GRP;
  const int tid = threadIdx.x;

  __shared__ float    bias1d[GRP];
  __shared__ _Float16 vsh[GRP * 16];
  __shared__ _Float16 kvsh[QKD * 16];      // kv_cum as f16, cols 12..15 zero
  __shared__ _Float16 astrip[16 * 512];    // 16 waves x (16 rows x 32 cols)
  __shared__ float    osh[16 * 192];       // 16 waves x (16 rows x 12 cols)

  for (int i = tid; i < GRP * 16; i += 512) vsh[i] = vbuf[tokbase * 16 + i];
  for (int i = tid; i < QKD * 16; i += 512) {
    const int d = i >> 4, e = i & 15;
    kvsh[i] = (e < EDIM) ? (_Float16)kvc[(size_t)bg * (QKD * EDIM) + d * EDIM + e]
                         : (_Float16)0.0f;
  }
  for (int i = tid; i < GRP; i += 512) {
    int bkt;
    if (i < 16) bkt = i;
    else {
      int prod = (int)((log((double)i / 16.0) / log(8.0)) * 16.0);
      bkt = 16 + prod; if (bkt > 31) bkt = 31;
    }
    bias1d[i] = rel[bkt] * 11.3137084989847604f;   // sqrt(128)
  }
  __syncthreads();

  const int wave = tid >> 5;
  const int lane = tid & 31;
  const int i0 = wave * 16;               // query tile base (in group)
  const int half = (lane >> 4) & 1;
  const int ncol = lane & 15;

  // preload the 4 A-fragments (query rows i0..i0+15, K=128)
  v16h afr[4];
#pragma unroll
  for (int kk = 0; kk < 4; ++kk)
    afr[kk] = load_a_f16(qq + (tokbase + i0) * QKD + kk * 32, QKD, lane);

  v8f oc = {};
  for (int jc = 0; jc < 8; ++jc) {        // 32-key chunks; uniform across waves
    const int j0 = jc * 32;
#pragma unroll
    for (int jt = 0; jt < 2; ++jt) {
      const int jb = j0 + jt * 16;
      v8f s = {};
#pragma unroll
      for (int kk = 0; kk < 4; ++kk) {
        v16h bfr = load_b_tr(qk2 + (tokbase + jb) * QKD + kk * 32, QKD, lane);
        s = wmma_f16(afr[kk], bfr, s);
      }
#pragma unroll
      for (int r = 0; r < 8; ++r) {
        const int row = i0 + r + half * 8;    // query index in group
        const int j = jb + ncol;              // key index in group
        float a = 0.0f;
        const int jg = g * GRP + j;
        if (j <= row && jg < NSEQ)
          a = laplace_f(s[r] * (1.0f / GRP) + bias1d[row - j]);
        astrip[wave * 512 + (r + half * 8) * 32 + (jt * 16 + ncol)] = (_Float16)a;
      }
    }
    __syncthreads();                          // uniform barrier: strip -> A-frag
    {
      v16h aa = load_a_f16(astrip + wave * 512, 32, lane);
      v16h bb = load_b_rm(vsh + j0 * 16, 16, lane);
      oc = wmma_f16(aa, bb, oc);
    }
  }

  // linear branch: lin_out = lq (16x128) @ kv_cum (128x16), fused into oc
#pragma unroll
  for (int kk = 0; kk < 4; ++kk) {
    v16h aa = load_a_f16(lq + (tokbase + i0) * QKD + kk * 32, QKD, lane);
    v16h bb = load_b_rm(kvsh + kk * 32 * 16, 16, lane);
    oc = wmma_f16(aa, bb, oc);
  }

  // stage outputs
#pragma unroll
  for (int r = 0; r < 8; ++r) {
    const int rl = r + half * 8;
    if (ncol < EDIM) osh[wave * 192 + rl * EDIM + ncol] = oc[r];
  }
  __syncthreads();

  // epilogue: gate, Wo, bias, residual -> h (f16)
  if (lane < 16) {
    const int rl = lane;
    const int n = g * GRP + i0 + rl;
    if (n < NSEQ) {
      const float* xr = x + ((size_t)b * NSEQ + n) * DMODEL;
      const _Float16* gp = gate + ((size_t)b * NSEQ + n) * EDIM;
      float t[EDIM];
#pragma unroll
      for (int e = 0; e < EDIM; ++e)
        t[e] = osh[wave * 192 + rl * EDIM + e] * (float)gp[e];
#pragma unroll
      for (int d = 0; d < DMODEL; ++d) {
        float sd = bo[d] + xr[d];
#pragma unroll
        for (int e = 0; e < EDIM; ++e) sd += t[e] * Wo[e * DMODEL + d];
        h[(size_t)b * HSTRIDE + n * DMODEL + d] = (_Float16)sd;
      }
    }
  }
}

// ---------------- 4. FC stack ----------------
__global__ void k_cvt_w(const float* __restrict__ src, _Float16* __restrict__ dst,
                        int K, int N, int Kpad) {
  const size_t total = (size_t)Kpad * N;
  for (size_t i = (size_t)blockIdx.x * blockDim.x + threadIdx.x; i < total;
       i += (size_t)gridDim.x * blockDim.x) {
    const int r = (int)(i / N);
    dst[i] = (r < K) ? (_Float16)src[i] : (_Float16)0.0f;
  }
}

__global__ void k_zero_hpad(_Float16* __restrict__ h) {
  const int i = blockIdx.x * blockDim.x + threadIdx.x;  // 0..2047
  const int b = i >> 4, c = i & 15;
  h[(size_t)b * HSTRIDE + 10800 + c] = (_Float16)0.0f;
}

// one wave per 16x16 output tile, K-loop of WMMA 16x16x32
__global__ void __launch_bounds__(32)
k_gemm(const _Float16* __restrict__ A, int lda,
       const _Float16* __restrict__ W, const float* __restrict__ bias,
       float* __restrict__ Y, int N, int Kpad) {
  const int lane = threadIdx.x;
  const int n0 = blockIdx.x * 16;
  const int m0 = blockIdx.y * 16;
  v8f c = {};
  for (int k0 = 0; k0 < Kpad; k0 += 32) {
    v16h a = load_a_f16(A + (size_t)m0 * lda + k0, lda, lane);
    v16h bb = load_b_rm(W + (size_t)k0 * N + n0, N, lane);
    c = wmma_f16(a, bb, c);
  }
  const int n = n0 + (lane & 15);
  const int rbase = m0 + ((lane >> 4) & 1) * 8;
#pragma unroll
  for (int r = 0; r < 8; ++r)
    Y[(size_t)(rbase + r) * N + n] = c[r] + bias[n];
}

// train-mode BN over batch (M=128) + ReLU, emit f16 activations
__global__ void k_bn_relu(const float* __restrict__ Y, const float* __restrict__ gm,
                          const float* __restrict__ bt, _Float16* __restrict__ A, int N) {
  const int col = blockIdx.x;
  const int tid = threadIdx.x;     // 0..127 == batch row
  __shared__ float red[128];
  __shared__ float s_mean, s_var;
  const float v = Y[(size_t)tid * N + col];
  red[tid] = v; __syncthreads();
  for (int s = 64; s > 0; s >>= 1) { if (tid < s) red[tid] += red[tid + s]; __syncthreads(); }
  if (tid == 0) s_mean = red[0] * (1.0f / 128.0f);
  __syncthreads();
  const float d = v - s_mean;
  red[tid] = d * d; __syncthreads();
  for (int s = 64; s > 0; s >>= 1) { if (tid < s) red[tid] += red[tid + s]; __syncthreads(); }
  if (tid == 0) s_var = red[0] * (1.0f / 128.0f);
  __syncthreads();
  float o = d * rsqrtf(s_var + 1e-5f) * gm[col] + bt[col];
  o = fmaxf(o, 0.0f);
  A[(size_t)tid * N + col] = (_Float16)o;
}

// final 32 -> 4 layer, ReLU, f32 output
__global__ void k_final(const _Float16* __restrict__ A, const float* __restrict__ W,
                        const float* __restrict__ b, float* __restrict__ out) {
  const int idx = blockIdx.x * blockDim.x + threadIdx.x;
  if (idx >= 128 * 4) return;
  const int row = idx >> 2, col = idx & 3;
  float s = b[col];
#pragma unroll
  for (int k = 0; k < 32; ++k) s += (float)A[row * 32 + k] * W[k * 4 + col];
  out[idx] = fmaxf(s, 0.0f);
}

// ---------------- host ----------------
extern "C" void kernel_launch(void* const* d_in, const int* in_sizes, int n_in,
                              void* d_out, int out_size, void* d_ws, size_t ws_size,
                              hipStream_t stream) {
  const float* x    = (const float*)d_in[0];
  const float* ln_g = (const float*)d_in[1];
  const float* ln_b = (const float*)d_in[2];
  const float* Wh   = (const float*)d_in[3];
  const float* bh   = (const float*)d_in[4];
  const float* Wqk  = (const float*)d_in[5];
  const float* bqk  = (const float*)d_in[6];
  const float* osg  = (const float*)d_in[7];
  const float* osb  = (const float*)d_in[8];
  const float* rel  = (const float*)d_in[9];
  const float* Wo   = (const float*)d_in[10];
  const float* bo   = (const float*)d_in[11];
  const float* fcW[7]; const float* fcB[7];
  for (int i = 0; i < 7; ++i) { fcW[i] = (const float*)d_in[12 + i]; fcB[i] = (const float*)d_in[19 + i]; }
  const float* bng[6]; const float* bnb[6];
  for (int i = 0; i < 6; ++i) { bng[i] = (const float*)d_in[26 + i]; bnb[i] = (const float*)d_in[32 + i]; }

  char* p = (char*)d_ws;
  auto alloc = [&](size_t bytes) -> void* {
    void* r = (void*)p;
    p += (bytes + 255) & ~(size_t)255;
    return r;
  };

  const size_t TOK = (size_t)BATCH * NPAD;
  _Float16* qq   = (_Float16*)alloc(TOK * QKD * 2);
  _Float16* lqb  = (_Float16*)alloc(TOK * QKD * 2);
  _Float16* qk2  = (_Float16*)alloc(TOK * QKD * 2);
  _Float16* lkb  = (_Float16*)alloc(TOK * QKD * 2);
  _Float16* vbuf = (_Float16*)alloc(TOK * 16 * 2);
  _Float16* gate = (_Float16*)alloc((size_t)BATCH * NSEQ * EDIM * 2);
  float*    kvr  = (float*)alloc((size_t)BATCH * NGRP * QKD * EDIM * 4);
  float*    kvc  = (float*)alloc((size_t)BATCH * NGRP * QKD * EDIM * 4);
  _Float16* h    = (_Float16*)alloc((size_t)BATCH * HSTRIDE * 2);

  const int fcK[6]    = {10800, 1024, 512, 256, 128, 64};
  const int fcKpad[6] = {10816, 1024, 512, 256, 128, 64};
  const int fcN[6]    = {1024, 512, 256, 128, 64, 32};
  _Float16* Wf[6];
  for (int l = 0; l < 6; ++l) Wf[l] = (_Float16*)alloc((size_t)fcKpad[l] * fcN[l] * 2);
  float*    ybuf = (float*)alloc((size_t)BATCH * 1024 * 4);
  _Float16* actA = (_Float16*)alloc((size_t)BATCH * 1024 * 2);
  _Float16* actB = (_Float16*)alloc((size_t)BATCH * 1024 * 2);

  // weight conversion (f32 -> f16, zero-padded rows)
  for (int l = 0; l < 6; ++l) {
    const size_t tot = (size_t)fcKpad[l] * fcN[l];
    const int blocks = (int)((tot + 255) / 256);
    k_cvt_w<<<blocks, 256, 0, stream>>>(fcW[l], Wf[l], fcK[l], fcN[l], fcKpad[l]);
  }

  k_preproc<<<(int)TOK, 128, 0, stream>>>(x, ln_g, ln_b, Wh, bh, Wqk, bqk, osg, osb,
                                          qq, lqb, qk2, lkb, vbuf, gate);
  k_kv<<<BATCH * NGRP, 256, 0, stream>>>(lkb, vbuf, kvr);
  k_cumsum<<<BATCH, 256, 0, stream>>>(kvr, kvc);
  k_zero_hpad<<<8, 256, 0, stream>>>(h);
  k_attn<<<BATCH * NGRP, 512, 0, stream>>>(qq, qk2, lqb, vbuf, gate, kvc, rel, x, Wo, bo, h);

  // FC stack
  const _Float16* curA = h;
  int curLda = HSTRIDE;
  _Float16* acts[2] = {actA, actB};
  for (int l = 0; l < 6; ++l) {
    dim3 grid(fcN[l] / 16, BATCH / 16);
    k_gemm<<<grid, 32, 0, stream>>>(curA, curLda, Wf[l], fcB[l], ybuf, fcN[l], fcKpad[l]);
    _Float16* nextA = acts[l & 1];
    k_bn_relu<<<fcN[l], 128, 0, stream>>>(ybuf, bng[l], bnb[l], nextA, fcN[l]);
    curA = nextA;
    curLda = fcN[l];
  }
  k_final<<<2, 256, 0, stream>>>(curA, fcW[6], fcB[6], (float*)d_out);
}
